// Dependency_GCNLayer_18098992185956
// MI455X (gfx1250) — compile-verified, matrix operand order checked by reading, not run on hardware
//
#include <hip/hip_runtime.h>
#include <hip/hip_bf16.h>

typedef __attribute__((ext_vector_type(16))) _Float16 v16h;
typedef __attribute__((ext_vector_type(8)))  float    v8f;

#define DIMD 256
#define AS_STRIDE 264   // 8-half pad: lane m -> bank 4*m + const (conflict-free), 528B row = 16B multiple

#ifndef __has_builtin
#define __has_builtin(x) 0
#endif
#if __has_builtin(__builtin_amdgcn_global_load_async_to_lds_b128)
#define USE_ASYNC_LDS 1
#else
#define USE_ASYNC_LDS 0
#endif

#if USE_ASYNC_LDS
typedef __attribute__((ext_vector_type(4))) int i4v;
typedef i4v __attribute__((address_space(1))) g_i4v;   // global (AS1) int4
typedef i4v __attribute__((address_space(3))) l_i4v;   // LDS    (AS3) int4

__device__ __forceinline__ l_i4v* to_lds(void* p) {
  // generic LDS pointer low 32 bits are the LDS byte offset (ISA 10.2)
  return (l_i4v*)(unsigned)(uintptr_t)p;
}
__device__ __forceinline__ g_i4v* to_gbl(const void* p) {
  return (g_i4v*)(uintptr_t)p;
}
__device__ __forceinline__ void wait_async0() {
#if __has_builtin(__builtin_amdgcn_s_wait_asynccnt)
  __builtin_amdgcn_s_wait_asynccnt(0);
#else
  asm volatile("s_wait_asynccnt 0x0" ::: "memory");
#endif
}
#endif

// ---- WMMA fragment helpers (CDNA5 ISA 7.12.2 16-bit layouts) ---------------

__device__ __forceinline__ int frag_kbase(int v, int hs) {
  return (v < 4) ? (2 * v + 8 * hs) : (16 + 2 * (v - 4) + 8 * hs);
}

__device__ __forceinline__ v16h load_a_lds(const _Float16* As, int lane, int k0) {
  int m = lane & 15, hs = (lane >> 4) & 1;
  const _Float16* row = As + m * AS_STRIDE + k0;
  v16h a;
#pragma unroll
  for (int v = 0; v < 8; ++v) {
    int kb = frag_kbase(v, hs);
    a[2 * v]     = row[kb];
    a[2 * v + 1] = row[kb + 1];
  }
  return a;
}

// B operand = rows of W (msg[i,o] = sum_k A[i,k]*W[o,k])
__device__ __forceinline__ v16h load_b_rows(const _Float16* W, int n0, int lane, int k0) {
  int n = lane & 15, hs = (lane >> 4) & 1;
  const _Float16* row = W + (size_t)(n0 + n) * DIMD + k0;
  v16h b;
#pragma unroll
  for (int v = 0; v < 8; ++v) {
    int kb = frag_kbase(v, hs);
    b[2 * v]     = row[kb];
    b[2 * v + 1] = row[kb + 1];
  }
  return b;
}

// ---- Phase 0: f32 -> f16 staging -------------------------------------------

__global__ void __launch_bounds__(256) convert_weights(
    const float* Wrel, const float* Wself, _Float16* WrelH, _Float16* WselfH,
    int nRel, int nSelf) {
  int i = blockIdx.x * blockDim.x + threadIdx.x;
  if (i < nRel)  WrelH[i]  = (_Float16)Wrel[i];
  if (i < nSelf) WselfH[i] = (_Float16)Wself[i];
}

__global__ void __launch_bounds__(256) convert_x(const float* x, _Float16* xH, int n) {
  int i = blockIdx.x * blockDim.x + threadIdx.x;
  if (i < n) xH[i] = (_Float16)x[i];
}

// ---- Phase 1: bucket edges by relation label (counting sort) ---------------

__global__ void __launch_bounds__(256) count_tasks(
    const int* dep, const int* rel, const int* gov, int E, int Rr, unsigned* counts) {
  int t = blockIdx.x * blockDim.x + threadIdx.x;
  if (t >= 2 * E) return;
  int e = (t < E) ? t : t - E;
  int w = rel[e] + ((t < E) ? 0 : Rr);
  atomicAdd(&counts[w], 1u);
}

__global__ void scan_buckets(const unsigned* counts, int NB, int* offs, int* tiles,
                             unsigned* cursor) {
  if (threadIdx.x == 0) {
    int acc = 0, tacc = 0;
    for (int b = 0; b < NB; ++b) {
      offs[b] = acc; tiles[b] = tacc; cursor[b] = (unsigned)acc;
      acc  += (int)counts[b];
      tacc += ((int)counts[b] + 31) / 32;   // 32 messages per tile
    }
    offs[NB] = acc; tiles[NB] = tacc;
  }
}

__global__ void __launch_bounds__(256) scatter_tasks(
    const int* dep, const int* rel, const int* gov, int E, int Rr,
    unsigned* cursor, int* srcS, int* dstS) {
  int t = blockIdx.x * blockDim.x + threadIdx.x;
  if (t >= 2 * E) return;
  int e, w, s, d;
  if (t < E) { e = t;     w = rel[e];      s = gov[e]; d = dep[e]; }   // forward
  else       { e = t - E; w = rel[e] + Rr; s = dep[e]; d = gov[e]; }   // reverse
  unsigned pos = atomicAdd(&cursor[w], 1u);
  srcS[pos] = s; dstS[pos] = d;
}

// ---- Phase 2: self transform, out = x @ W_self^T + b_self (WMMA, M=32) -----

__global__ void __launch_bounds__(256) self_gemm(
    const _Float16* xH, const _Float16* Wh, const float* bias, float* out, int Nn) {
  __shared__ __align__(16) _Float16 As[32 * AS_STRIDE];
  int tid = threadIdx.x;
  int wave = tid >> 5, lane = tid & 31;
  int row0 = blockIdx.x * 32;

#if USE_ASYNC_LDS
#pragma unroll
  for (int rr = 0; rr < 4; ++rr) {              // 8 waves x 4 rows = 32 rows
    int r = wave * 4 + rr;
    int s = row0 + r; if (s >= Nn) s = Nn - 1;  // clamp; extra rows discarded at store
    __builtin_amdgcn_global_load_async_to_lds_b128(
        to_gbl(xH + (size_t)s * DIMD + lane * 8),
        to_lds(As + r * AS_STRIDE + lane * 8), 0, 0);
  }
  wait_async0();
#else
  for (int i = tid; i < 32 * (DIMD / 8); i += 256) {
    int r = i >> 5, c8 = i & 31;
    int s = row0 + r; if (s >= Nn) s = Nn - 1;
    *(uint4*)(As + r * AS_STRIDE + c8 * 8) =
        *(const uint4*)(xH + (size_t)s * DIMD + c8 * 8);
  }
#endif
  __syncthreads();

  int n0 = wave * 32;                 // 8 waves x 2 N-tiles cover all 256 outputs
  v8f c00 = {}, c01 = {}, c10 = {}, c11 = {};
#pragma unroll
  for (int k0 = 0; k0 < DIMD; k0 += 32) {
    v16h a0 = load_a_lds(As, lane, k0);
    v16h a1 = load_a_lds(As + 16 * AS_STRIDE, lane, k0);
    v16h b0 = load_b_rows(Wh, n0,      lane, k0);
    v16h b1 = load_b_rows(Wh, n0 + 16, lane, k0);
    c00 = __builtin_amdgcn_wmma_f32_16x16x32_f16(false, a0, false, b0, (short)0, c00, false, false);
    c01 = __builtin_amdgcn_wmma_f32_16x16x32_f16(false, a0, false, b1, (short)0, c01, false, false);
    c10 = __builtin_amdgcn_wmma_f32_16x16x32_f16(false, a1, false, b0, (short)0, c10, false, false);
    c11 = __builtin_amdgcn_wmma_f32_16x16x32_f16(false, a1, false, b1, (short)0, c11, false, false);
  }
  int col = lane & 15, hs = lane >> 4;
  float bi0 = bias[n0 + col], bi1 = bias[n0 + 16 + col];
#pragma unroll
  for (int r = 0; r < 8; ++r) {
    int m = r + 8 * hs;
    int g0 = row0 + m, g1 = row0 + 16 + m;
    if (g0 < Nn) {
      out[(size_t)g0 * DIMD + n0 + col]      = c00[r] + bi0;
      out[(size_t)g0 * DIMD + n0 + 16 + col] = c01[r] + bi1;
    }
    if (g1 < Nn) {
      out[(size_t)g1 * DIMD + n0 + col]      = c10[r] + bi0;
      out[(size_t)g1 * DIMD + n0 + 16 + col] = c11[r] + bi1;
    }
  }
}

// ---- Phase 3: relation-grouped message GEMM + scatter-add (WMMA, M=32) -----

__global__ void __launch_bounds__(256) msg_gemm(
    const _Float16* xH, const _Float16* WrelH, const float* brel,
    const int* srcS, const int* dstS, const int* offs, const int* tiles, int NB,
    float* out) {
  __shared__ __align__(16) _Float16 As[32 * AS_STRIDE];
  __shared__ int sSrc[32], sDst[32];
  __shared__ int sB;
  int tid = threadIdx.x;
  int wave = tid >> 5, lane = tid & 31;
  int totalTiles = tiles[NB];

  for (int t = blockIdx.x; t < totalTiles; t += gridDim.x) {
    if (tid == 0) {                       // tiles[] monotone; skips empty buckets
      int b = 0;
      while (b + 1 < NB && t >= tiles[b + 1]) ++b;
      sB = b;
    }
    __syncthreads();
    int b = sB;
    int base = offs[b] + (t - tiles[b]) * 32;
    int lim  = offs[b + 1];
    if (tid < 32) {
      int idx = base + tid;
      if (idx < lim) { sSrc[tid] = srcS[idx]; sDst[tid] = dstS[idx]; }
      else           { sSrc[tid] = 0;         sDst[tid] = -1; }   // safe pad row
    }
    const _Float16* Wb = WrelH + (size_t)b * DIMD * DIMD;
    __builtin_prefetch(Wb + (size_t)tid * DIMD, 0, 3);   // global_prefetch_b8
    __syncthreads();

#if USE_ASYNC_LDS
#pragma unroll
    for (int rr = 0; rr < 4; ++rr) {
      int r = wave * 4 + rr;
      int s = sSrc[r];
      __builtin_amdgcn_global_load_async_to_lds_b128(
          to_gbl(xH + (size_t)s * DIMD + lane * 8),
          to_lds(As + r * AS_STRIDE + lane * 8), 0, 0);
    }
    wait_async0();
#else
    for (int i = tid; i < 32 * (DIMD / 8); i += 256) {
      int r = i >> 5, c8 = i & 31;
      int s = sSrc[r];
      *(uint4*)(As + r * AS_STRIDE + c8 * 8) =
          *(const uint4*)(xH + (size_t)s * DIMD + c8 * 8);
    }
#endif
    __syncthreads();

    int n0 = wave * 32;
    v8f c00 = {}, c01 = {}, c10 = {}, c11 = {};
#pragma unroll
    for (int k0 = 0; k0 < DIMD; k0 += 32) {
      v16h a0 = load_a_lds(As, lane, k0);
      v16h a1 = load_a_lds(As + 16 * AS_STRIDE, lane, k0);
      v16h b0 = load_b_rows(Wb, n0,      lane, k0);
      v16h b1 = load_b_rows(Wb, n0 + 16, lane, k0);
      c00 = __builtin_amdgcn_wmma_f32_16x16x32_f16(false, a0, false, b0, (short)0, c00, false, false);
      c01 = __builtin_amdgcn_wmma_f32_16x16x32_f16(false, a0, false, b1, (short)0, c01, false, false);
      c10 = __builtin_amdgcn_wmma_f32_16x16x32_f16(false, a1, false, b0, (short)0, c10, false, false);
      c11 = __builtin_amdgcn_wmma_f32_16x16x32_f16(false, a1, false, b1, (short)0, c11, false, false);
    }

    int col = lane & 15, hs = lane >> 4;
    float bb0 = brel[(size_t)b * DIMD + n0 + col];
    float bb1 = brel[(size_t)b * DIMD + n0 + 16 + col];
#pragma unroll
    for (int r = 0; r < 8; ++r) {
      int m = r + 8 * hs;
      int d0 = sDst[m], d1 = sDst[16 + m];
      if (d0 >= 0) {
        atomicAdd(&out[(size_t)d0 * DIMD + n0 + col],      c00[r] + bb0);
        atomicAdd(&out[(size_t)d0 * DIMD + n0 + 16 + col], c01[r] + bb1);
      }
      if (d1 >= 0) {
        atomicAdd(&out[(size_t)d1 * DIMD + n0 + col],      c10[r] + bb0);
        atomicAdd(&out[(size_t)d1 * DIMD + n0 + 16 + col], c11[r] + bb1);
      }
    }
    __syncthreads();
  }
}

// ---- Phase 4: ReLU ---------------------------------------------------------

__global__ void __launch_bounds__(256) relu_kernel(float* out, int n) {
  int i = blockIdx.x * blockDim.x + threadIdx.x;
  if (i < n) out[i] = fmaxf(out[i], 0.f);
}

// ---- Host-side orchestration (stream-ordered, capture-safe) ----------------

extern "C" void kernel_launch(void* const* d_in, const int* in_sizes, int n_in,
                              void* d_out, int out_size, void* d_ws, size_t ws_size,
                              hipStream_t stream) {
  const float* x     = (const float*)d_in[0];
  const int*   dep   = (const int*)d_in[1];
  const int*   rel   = (const int*)d_in[2];
  const int*   gov   = (const int*)d_in[3];
  const float* Wself = (const float*)d_in[4];
  const float* bself = (const float*)d_in[5];
  const float* Wrel  = (const float*)d_in[6];
  const float* brel  = (const float*)d_in[7];
  float* out = (float*)d_out;

  const int D    = DIMD;
  const int Nn   = in_sizes[0] / D;
  const int E    = in_sizes[1];
  const int twoR = in_sizes[6] / (D * D);
  const int Rr   = twoR / 2;

  char* ws = (char*)d_ws;
  unsigned* counts  = (unsigned*)(ws);         // [twoR]
  unsigned* cursor  = (unsigned*)(ws + 128);   // [twoR]
  int*      offsA   = (int*)(ws + 256);        // [twoR+1]
  int*      tilesA  = (int*)(ws + 384);        // [twoR+1]
  size_t o = 512;
  int* srcS = (int*)(ws + o); o += (size_t)2 * E * sizeof(int);
  int* dstS = (int*)(ws + o); o += (size_t)2 * E * sizeof(int);
  _Float16* WrelH  = (_Float16*)(ws + o); o += (size_t)twoR * D * D * sizeof(_Float16);
  _Float16* WselfH = (_Float16*)(ws + o); o += (size_t)D * D * sizeof(_Float16);
  _Float16* xH     = (_Float16*)(ws + o); o += (size_t)Nn * D * sizeof(_Float16);

  (void)hipMemsetAsync(ws, 0, 128, stream);  // zero bucket counters every call

  int nRel = twoR * D * D, nSelf = D * D, nX = Nn * D;
  convert_weights<<<(nRel + 255) / 256, 256, 0, stream>>>(Wrel, Wself, WrelH, WselfH, nRel, nSelf);
  convert_x<<<(nX + 255) / 256, 256, 0, stream>>>(x, xH, nX);
  count_tasks<<<(2 * E + 255) / 256, 256, 0, stream>>>(dep, rel, gov, E, Rr, counts);
  scan_buckets<<<1, 32, 0, stream>>>(counts, twoR, offsA, tilesA, cursor);
  scatter_tasks<<<(2 * E + 255) / 256, 256, 0, stream>>>(dep, rel, gov, E, Rr, cursor, srcS, dstS);
  self_gemm<<<(Nn + 31) / 32, 256, 0, stream>>>(xH, WselfH, bself, out, Nn);
  msg_gemm<<<1024, 256, 0, stream>>>(xH, WrelH, brel, srcS, dstS, offsA, tilesA, twoR, out);
  relu_kernel<<<((size_t)Nn * D + 255) / 256, 256, 0, stream>>>(out, Nn * D);
}
